// AttentionLayer_84593675862192
// MI455X (gfx1250) — matrix operand order repaired and checked by reading, not compiled, not run on hardware
//
#include <hip/hip_runtime.h>

// ---- CDNA5 WMMA types ----
typedef unsigned short u16;
typedef unsigned int u32;
typedef __attribute__((ext_vector_type(16))) __bf16 v16bf;
typedef __attribute__((ext_vector_type(8))) float v8f;

union FragU { uint4 u[2]; v16bf v; };

__device__ inline u16 f2bf(float f) {
  u32 u = __float_as_uint(f);
  u32 r = u + 0x7FFFu + ((u >> 16) & 1u);   // round-to-nearest-even
  return (u16)(r >> 16);
}

__device__ inline v8f wmma_bf16(v16bf a, v16bf b, v8f c) {
  return __builtin_amdgcn_wmma_f32_16x16x32_bf16(false, a, false, b, (short)0, c,
                                                 false, false);
}

// ---- CDNA5 async global->LDS copy (ASYNCcnt path), with sync fallback ----
#if __has_builtin(__builtin_amdgcn_global_load_async_to_lds_b128) && \
    __has_builtin(__builtin_amdgcn_s_wait_asynccnt)
#define USE_ASYNC_LDS 1
#else
#define USE_ASYNC_LDS 0
#endif

// Builtin expects: (int4 __device__*, int4 __shared__*, imm offset, imm cpol)
// with int4 = __attribute__((vector_size(16))) int  (per hipcc diagnostic).
typedef int v4i_gcc __attribute__((vector_size(16)));
typedef __attribute__((address_space(1))) v4i_gcc* as1_v4ip;
typedef __attribute__((address_space(3))) v4i_gcc* as3_v4ip;

__device__ inline void copy16_to_lds(const u16* g, u16* l) {
#if USE_ASYNC_LDS
  __builtin_amdgcn_global_load_async_to_lds_b128((as1_v4ip)(void*)g,
                                                 (as3_v4ip)(void*)l, 0, 0);
#else
  *(uint4*)l = *(const uint4*)g;
#endif
}

__device__ inline void wait_async_lds() {
#if USE_ASYNC_LDS
  __builtin_amdgcn_s_wait_asynccnt(0);
#endif
}

// Problem constants (reference: B=2, S=2048, D=1024, H=16, DK=64, DF=4096)
#define C_B 2
#define C_S 2048
#define C_D 1024
#define C_H 16
#define C_DK 64
#define C_DF 4096
#define C_MT (C_B * C_S)   // 4096 token rows

// ---------------- f32 -> bf16 elementwise convert ----------------
__global__ void cvt_kernel(const float* __restrict__ in, u16* __restrict__ out, int n) {
  int i = blockIdx.x * blockDim.x + threadIdx.x;
  if (i < n) out[i] = f2bf(in[i]);
}

// ---------------- LayerNorm (f32 in, bf16 out), one block per row ----------------
__global__ __launch_bounds__(256) void ln_kernel(const float* __restrict__ x,
                                                 const float* __restrict__ g,
                                                 const float* __restrict__ be,
                                                 u16* __restrict__ out) {
  int row = blockIdx.x;
  int tid = threadIdx.x;
  const float* xr = x + (size_t)row * C_D;
  float s = 0.f, ss = 0.f;
#pragma unroll
  for (int i = 0; i < 4; ++i) {
    float v = xr[tid + i * 256];
    s += v;
    ss += v * v;
  }
#pragma unroll
  for (int off = 16; off; off >>= 1) {
    s += __shfl_xor(s, off, 32);
    ss += __shfl_xor(ss, off, 32);
  }
  __shared__ float sh[2][8];
  if ((tid & 31) == 0) { sh[0][tid >> 5] = s; sh[1][tid >> 5] = ss; }
  __syncthreads();
  s = 0.f; ss = 0.f;
#pragma unroll
  for (int w = 0; w < 8; ++w) { s += sh[0][w]; ss += sh[1][w]; }
  float mu = s * (1.f / (float)C_D);
  float var = ss * (1.f / (float)C_D) - mu * mu;
  float rstd = rsqrtf(var + 1e-5f);
  u16* orow = out + (size_t)row * C_D;
#pragma unroll
  for (int i = 0; i < 4; ++i) {
    int c = tid + i * 256;
    orow[c] = f2bf((xr[c] - mu) * rstd * g[c] + be[c]);
  }
}

// ---------------- bf16 GEMM: C[M,N] = A[M,K] * W[K,N] ----------------
// 256 threads = 8 waves; block tile 128(M) x 128(N); wave tile 32x64 (2x4 WMMA).
// Double-buffered LDS; A tile staged with async global->LDS DMA (ASYNCcnt).
// mode 0: f32 out = acc + bias[n] + res[m,n]
// mode 1: bf16 out = relu(acc + bias[n]), row-major
// mode 2: bf16 scatter to [B,H,S,DK]   (Q, K)
// mode 3: bf16 scatter to [B,H,DK,S]   (V transposed)
__global__ __launch_bounds__(256) void gemm_bf16_kernel(
    const u16* __restrict__ A, const u16* __restrict__ W, int M, int N, int K,
    int mode, const float* __restrict__ bias, const float* __restrict__ res,
    float* __restrict__ outF, u16* __restrict__ outB) {
  __shared__ __attribute__((aligned(16))) u16 As[2][128][40];  // row-major, 10KB/buf
  __shared__ __attribute__((aligned(16))) u16 Bs[2][128][48];  // TRANSPOSED [n][k], 12KB/buf

  int tid = threadIdx.x;
  int lane = tid & 31, half = lane >> 4, l16 = lane & 15;
  int wave = tid >> 5;
  int wm = (wave & 3) * 32, wn = (wave >> 2) * 64;
  int gm0 = blockIdx.y * 128, gn0 = blockIdx.x * 128;

  int ar = tid >> 1, ac = (tid & 1) * 16;  // A tile: 128 rows x 32 k, 32B/thread
  int kr = tid >> 3, nc = (tid & 7) * 16;  // W tile: 32 k-rows x 128 n, 32B/thread

  const u16* Arow = A + (size_t)(gm0 + ar) * K + ac;
  const u16* Brow = W + (size_t)kr * N + gn0 + nc;

  v8f acc[2][4];
#pragma unroll
  for (int i = 0; i < 2; ++i)
#pragma unroll
    for (int j = 0; j < 4; ++j) {
      v8f z = {0.f, 0.f, 0.f, 0.f, 0.f, 0.f, 0.f, 0.f};
      acc[i][j] = z;
    }

  int nk = K / 32;

  // ---- prologue: stage tile 0 into buffer 0 ----
  {
    copy16_to_lds(Arow, &As[0][ar][ac]);
    copy16_to_lds(Arow + 8, &As[0][ar][ac + 8]);
    uint4 b0 = *(const uint4*)Brow;
    uint4 b1 = *(const uint4*)(Brow + 8);
    const u16* bw0 = (const u16*)&b0;
    const u16* bw1 = (const u16*)&b1;
#pragma unroll
    for (int i = 0; i < 8; ++i) {
      Bs[0][nc + i][kr] = bw0[i];
      Bs[0][nc + 8 + i][kr] = bw1[i];
    }
    wait_async_lds();
  }
  __syncthreads();

  for (int kt = 0; kt < nk; ++kt) {
    int p = kt & 1;
    bool nxt = (kt + 1) < nk;
    uint4 b0 = {}, b1 = {};
    if (nxt) {  // kick next A-tile DMA + fetch next B tile into regs
      const u16* An = Arow + (size_t)(kt + 1) * 32;
      copy16_to_lds(An, &As[1 - p][ar][ac]);
      copy16_to_lds(An + 8, &As[1 - p][ar][ac + 8]);
      const u16* Bn = Brow + (size_t)(kt + 1) * 32 * N;
      b0 = *(const uint4*)Bn;
      b1 = *(const uint4*)(Bn + 8);
    }

    // ---- compute from buffer p: 2 A frags x 4 B frags -> 8 WMMAs ----
    FragU fa[2], fb[4];
#pragma unroll
    for (int i = 0; i < 2; ++i) {
      const u16* pA = &As[p][wm + i * 16 + l16][half * 8];
      fa[i].u[0] = *(const uint4*)pA;
      fa[i].u[1] = *(const uint4*)(pA + 16);
    }
#pragma unroll
    for (int j = 0; j < 4; ++j) {
      const u16* pB = &Bs[p][wn + j * 16 + l16][half * 8];
      fb[j].u[0] = *(const uint4*)pB;
      fb[j].u[1] = *(const uint4*)(pB + 16);
    }
#pragma unroll
    for (int i = 0; i < 2; ++i)
#pragma unroll
      for (int j = 0; j < 4; ++j) acc[i][j] = wmma_bf16(fa[i].v, fb[j].v, acc[i][j]);

    if (nxt) {  // transpose-store next B tile into the other buffer
      const u16* bw0 = (const u16*)&b0;
      const u16* bw1 = (const u16*)&b1;
#pragma unroll
      for (int i = 0; i < 8; ++i) {
        Bs[1 - p][nc + i][kr] = bw0[i];
        Bs[1 - p][nc + 8 + i][kr] = bw1[i];
      }
      wait_async_lds();
    }
    __syncthreads();
  }

  // ---- epilogue. C layout: lane = col n=l16(+16j), rows m = v + half*8 ----
#pragma unroll
  for (int i = 0; i < 2; ++i)
#pragma unroll
    for (int j = 0; j < 4; ++j) {
      int n = gn0 + wn + j * 16 + l16;
#pragma unroll
      for (int v = 0; v < 8; ++v) {
        int m = gm0 + wm + i * 16 + half * 8 + v;
        float val = acc[i][j][v];
        if (mode == 0) {
          val += bias[n] + res[(size_t)m * N + n];
          outF[(size_t)m * N + n] = val;
        } else if (mode == 1) {
          val += bias[n];
          val = val > 0.f ? val : 0.f;
          outB[(size_t)m * N + n] = f2bf(val);
        } else {
          int bb = m >> 11, s = m & (C_S - 1);
          int h = n >> 6, dk = n & (C_DK - 1);
          size_t idx = (mode == 2)
                           ? (((size_t)(bb * C_H + h) * C_S + s) * C_DK + dk)
                           : (((size_t)(bb * C_H + h) * C_DK + dk) * C_S + s);
          outB[idx] = f2bf(val);
        }
      }
    }
}

// ---------------- Flash attention: one wave per 16-query tile ----------------
// Q,K: bf16 [B,H,S,DK]; Vt: bf16 [B,H,DK,S]; ctx out: bf16 [B,S,D] (concat).
__global__ __launch_bounds__(256) void attn_kernel(const u16* __restrict__ Q,
                                                   const u16* __restrict__ Kb,
                                                   const u16* __restrict__ Vt,
                                                   const int* __restrict__ mask,
                                                   u16* __restrict__ ctx) {
  __shared__ __attribute__((aligned(16))) u16 P[8][16][32];
  int tid = threadIdx.x;
  int lane = tid & 31, half = lane >> 4, l16 = lane & 15;
  int ws = tid >> 5;
  int wid = blockIdx.x * 8 + ws;
  int qt = wid & 127;          // S/16 = 128 query tiles
  int h = (wid >> 7) & 15;
  int b = wid >> 11;

  const u16* qp = Q + ((size_t)(b * C_H + h) * C_S + qt * 16 + l16) * C_DK;
  FragU qa0, qa1;
  qa0.u[0] = *(const uint4*)(qp + half * 8);
  qa0.u[1] = *(const uint4*)(qp + half * 8 + 16);
  qa1.u[0] = *(const uint4*)(qp + 32 + half * 8);
  qa1.u[1] = *(const uint4*)(qp + 32 + half * 8 + 16);

  float mrow[8], lrow[8];
#pragma unroll
  for (int v = 0; v < 8; ++v) { mrow[v] = -3.0e38f; lrow[v] = 0.f; }
  v8f o[4];
#pragma unroll
  for (int t = 0; t < 4; ++t) {
    v8f z = {0.f, 0.f, 0.f, 0.f, 0.f, 0.f, 0.f, 0.f};
    o[t] = z;
  }

  const u16* kbase = Kb + (size_t)(b * C_H + h) * C_S * C_DK;
  const u16* vbase = Vt + (size_t)(b * C_H + h) * C_DK * C_S;
  const int* mbase = mask + (size_t)b * C_S * C_S + (size_t)(qt * 16) * C_S;

  for (int jb = 0; jb < C_S; jb += 32) {
    v8f sc[2];
#pragma unroll
    for (int c = 0; c < 2; ++c) {
      v8f z = {0.f, 0.f, 0.f, 0.f, 0.f, 0.f, 0.f, 0.f};
      FragU k0f, k1f;
      const u16* kr = kbase + (size_t)(jb + c * 16 + l16) * C_DK + half * 8;
      k0f.u[0] = *(const uint4*)kr;
      k0f.u[1] = *(const uint4*)(kr + 16);
      k1f.u[0] = *(const uint4*)(kr + 32);
      k1f.u[1] = *(const uint4*)(kr + 48);
      z = wmma_bf16(qa0.v, k0f.v, z);
      sc[c] = wmma_bf16(qa1.v, k1f.v, z);
    }

#pragma unroll
    for (int v = 0; v < 8; ++v) {
      int qrow = half * 8 + v;
      const int* mp = mbase + (size_t)qrow * C_S + jb;
      float x0 = sc[0][v] * 0.125f;  // 1/sqrt(64)
      float x1 = sc[1][v] * 0.125f;
      if (mp[l16] == 0) x0 = -1e9f;
      if (mp[16 + l16] == 0) x1 = -1e9f;
      float mx = fmaxf(x0, x1);
#pragma unroll
      for (int off = 8; off; off >>= 1) mx = fmaxf(mx, __shfl_xor(mx, off, 32));
      float mnew = fmaxf(mrow[v], mx);
      float corr = __expf(mrow[v] - mnew);
      float p0 = __expf(x0 - mnew);
      float p1 = __expf(x1 - mnew);
      float rs = p0 + p1;
#pragma unroll
      for (int off = 8; off; off >>= 1) rs += __shfl_xor(rs, off, 32);
      lrow[v] = lrow[v] * corr + rs;
      mrow[v] = mnew;
      sc[0][v] = p0;
      sc[1][v] = p1;
#pragma unroll
      for (int t = 0; t < 4; ++t) o[t][v] = o[t][v] * corr;
    }

    // C-layout -> A-layout via per-wave LDS tile
#pragma unroll
    for (int v = 0; v < 8; ++v) {
      P[ws][half * 8 + v][l16] = f2bf(sc[0][v]);
      P[ws][half * 8 + v][16 + l16] = f2bf(sc[1][v]);
    }
    __syncthreads();
    FragU pa;
    const u16* pr = &P[ws][l16][0];
    pa.u[0] = *(const uint4*)(pr + half * 8);
    pa.u[1] = *(const uint4*)(pr + half * 8 + 16);
    __syncthreads();

#pragma unroll
    for (int t = 0; t < 4; ++t) {
      FragU vb;
      const u16* vr = vbase + (size_t)(t * 16 + l16) * C_S + jb + half * 8;
      vb.u[0] = *(const uint4*)vr;
      vb.u[1] = *(const uint4*)(vr + 16);
      o[t] = wmma_bf16(pa.v, vb.v, o[t]);
    }
  }

#pragma unroll
  for (int v = 0; v < 8; ++v) {
    float inv = 1.f / lrow[v];
    int s = qt * 16 + half * 8 + v;
    u16* cp = ctx + (size_t)(b * C_S + s) * C_D + h * C_DK;
#pragma unroll
    for (int t = 0; t < 4; ++t) cp[t * 16 + l16] = f2bf(o[t][v] * inv);
  }
}

extern "C" void kernel_launch(void* const* d_in, const int* in_sizes, int n_in,
                              void* d_out, int out_size, void* d_ws, size_t ws_size,
                              hipStream_t stream) {
  (void)in_sizes; (void)n_in; (void)out_size; (void)ws_size;
  const float* x = (const float*)d_in[0];
  const int* mask = (const int*)d_in[1];
  const float* Wq = (const float*)d_in[2];
  const float* Wk = (const float*)d_in[3];
  const float* Wv = (const float*)d_in[4];
  const float* Wo = (const float*)d_in[5];
  const float* bo = (const float*)d_in[6];
  const float* g1 = (const float*)d_in[7];
  const float* be1 = (const float*)d_in[8];
  const float* g2 = (const float*)d_in[9];
  const float* be2 = (const float*)d_in[10];
  const float* W1 = (const float*)d_in[11];
  const float* b1 = (const float*)d_in[12];
  const float* W2 = (const float*)d_in[13];
  const float* b2 = (const float*)d_in[14];
  float* out = (float*)d_out;

  char* ws = (char*)d_ws;
  size_t off = 0;
  auto alloc = [&](size_t bytes) {
    char* p = ws + off;
    off += (bytes + 255) & ~(size_t)255;
    return p;
  };
  u16* xa = (u16*)alloc((size_t)C_MT * C_D * 2);
  u16* wqb = (u16*)alloc((size_t)C_D * C_D * 2);
  u16* wkb = (u16*)alloc((size_t)C_D * C_D * 2);
  u16* wvb = (u16*)alloc((size_t)C_D * C_D * 2);
  u16* wob = (u16*)alloc((size_t)C_D * C_D * 2);
  u16* w1b = (u16*)alloc((size_t)C_D * C_DF * 2);
  u16* w2b = (u16*)alloc((size_t)C_DF * C_D * 2);
  u16* q = (u16*)alloc((size_t)C_MT * C_D * 2);
  u16* kk = (u16*)alloc((size_t)C_MT * C_D * 2);
  u16* vt = (u16*)alloc((size_t)C_MT * C_D * 2);
  u16* ctx = (u16*)alloc((size_t)C_MT * C_D * 2);
  float* x1 = (float*)alloc((size_t)C_MT * C_D * 4);
  u16* xf = (u16*)alloc((size_t)C_MT * C_D * 2);
  u16* h1 = (u16*)alloc((size_t)C_MT * C_DF * 2);

  // 1) weights -> bf16
  int nDD = C_D * C_D, nDF = C_D * C_DF;
  cvt_kernel<<<(nDD + 255) / 256, 256, 0, stream>>>(Wq, wqb, nDD);
  cvt_kernel<<<(nDD + 255) / 256, 256, 0, stream>>>(Wk, wkb, nDD);
  cvt_kernel<<<(nDD + 255) / 256, 256, 0, stream>>>(Wv, wvb, nDD);
  cvt_kernel<<<(nDD + 255) / 256, 256, 0, stream>>>(Wo, wob, nDD);
  cvt_kernel<<<(nDF + 255) / 256, 256, 0, stream>>>(W1, w1b, nDF);
  cvt_kernel<<<(nDF + 255) / 256, 256, 0, stream>>>(W2, w2b, nDF);

  // 2) LN1
  ln_kernel<<<C_MT, 256, 0, stream>>>(x, g1, be1, xa);

  // 3) QKV projections (scatter to head layouts, V transposed)
  dim3 gD(C_D / 128, C_MT / 128);  // (8, 32)
  gemm_bf16_kernel<<<gD, 256, 0, stream>>>(xa, wqb, C_MT, C_D, C_D, 2, nullptr,
                                           nullptr, nullptr, q);
  gemm_bf16_kernel<<<gD, 256, 0, stream>>>(xa, wkb, C_MT, C_D, C_D, 2, nullptr,
                                           nullptr, nullptr, kk);
  gemm_bf16_kernel<<<gD, 256, 0, stream>>>(xa, wvb, C_MT, C_D, C_D, 3, nullptr,
                                           nullptr, nullptr, vt);

  // 4) flash attention: B*H*(S/16)=4096 waves, 8 waves/block
  attn_kernel<<<512, 256, 0, stream>>>(q, kk, vt, mask, ctx);

  // 5) output projection + residual: x1 = x + ctx@Wo + bo
  gemm_bf16_kernel<<<gD, 256, 0, stream>>>(ctx, wob, C_MT, C_D, C_D, 0, bo, x, x1,
                                           nullptr);

  // 6) LN2
  ln_kernel<<<C_MT, 256, 0, stream>>>(x1, g2, be2, xf);

  // 7) FFN1 with ReLU -> bf16 h1
  dim3 gF(C_DF / 128, C_MT / 128);  // (32, 32)
  gemm_bf16_kernel<<<gF, 256, 0, stream>>>(xf, w1b, C_MT, C_DF, C_D, 1, b1, nullptr,
                                           nullptr, h1);

  // 8) FFN2 + residual -> final f32 output
  gemm_bf16_kernel<<<gD, 256, 0, stream>>>(h1, w2b, C_MT, C_D, C_DF, 0, b2, x1, out,
                                           nullptr);
}
